// GripNetInternalModule_71932112273423
// MI455X (gfx1250) — matrix operand order repaired and checked by reading, not compiled, not run on hardware
//
#include <hip/hip_runtime.h>
#include <stdint.h>

// RGCN (GripNet internal module) fused pipeline for MI455X (gfx1250).
// Edges are sorted by relation in equal blocks; per relation, gather 16
// source-node bf16 feature rows, multiply by the 32x32 relation weight with
// v_wmma_f32_16x16x32_bf16 (one WMMA per 16-col output tile, K=32 == full
// reduction), scatter-add fp32 results into an L2-resident accumulator using
// saddr-form global atomics (uniform base + u32 voffset).

#define NNODES 100000
#define NEDGES 2560000
#define NREL   8
#define NBASE  8
#define DIN    32
#define DOUT   32
#define OUTDIM 96

typedef __attribute__((ext_vector_type(16))) __bf16 v16bf;
typedef __attribute__((ext_vector_type(8)))  float  v8f;

static __device__ __forceinline__ uint16_t f2bf(float f) {
  union { float f; uint32_t u; } x; x.f = f;
  uint32_t r = x.u + 0x7FFFu + ((x.u >> 16) & 1u);   // round-to-nearest-even
  return (uint16_t)(r >> 16);
}

// ---- degree: one atomic per edge (done once; same graph both layers) -------
__global__ void deg_kernel(const int* __restrict__ dst, float* __restrict__ deg) {
  int i = blockIdx.x * blockDim.x + threadIdx.x;
  if (i < NEDGES) atomicAdd(&deg[(uint32_t)dst[i]], 1.0f);
}

// ---- copy x into out[:, 0:32] and produce bf16 feature table ---------------
__global__ void copy_convert_kernel(const float* __restrict__ x,
                                    float* __restrict__ out,
                                    uint16_t* __restrict__ xb) {
  int i = blockIdx.x * blockDim.x + threadIdx.x;
  if (i < NNODES * DIN) {
    int node = i >> 5, n = i & 31;
    float v = x[i];
    out[node * OUTDIM + n] = v;
    xb[i] = f2bf(v);
  }
}

// ---- w_r = sum_b att[r,b] * basis[b]  ->  bf16, transposed [R][n][k] -------
__global__ void wmat_kernel(const float* __restrict__ att,
                            const float* __restrict__ basis,
                            uint16_t* __restrict__ wb) {
  int r = blockIdx.x;
  int t = threadIdx.x;                 // 0..1023 -> (k, n)
  float acc = 0.0f;
#pragma unroll
  for (int b = 0; b < NBASE; ++b) acc += att[r * NBASE + b] * basis[b * 1024 + t];
  wb[r * 1024 + (t & 31) * 32 + (t >> 5)] = f2bf(acc);
}

// ---- fused gather -> WMMA -> scatter-add -----------------------------------
// grid = (blocks_x, NREL); 256 threads = 8 independent wave32's per block.
// Each wave processes 16-edge tiles of its relation block with a grid stride.
__global__ __launch_bounds__(256) void edge_wmma_kernel(
    const uint16_t* __restrict__ xb,   // [N][32] bf16 features
    const uint16_t* __restrict__ wb,   // [R][32][32] bf16 weights
    const int* __restrict__ src,
    const int* __restrict__ dst,
    const int* __restrict__ range_list,
    float* __restrict__ acc) {
  const int r    = blockIdx.y;
  const int lane = threadIdx.x & 31;
  const int wavesPerRel = gridDim.x * (blockDim.x >> 5);
  const int waveId      = blockIdx.x * (blockDim.x >> 5) + (threadIdx.x >> 5);

  const int e0 = range_list[2 * r];
  const int e1 = range_list[2 * r + 1];
  const int nTiles = (e1 - e0) >> 4;       // 16 edges per tile (E_PER % 16 == 0)

  // B operand (ISA 16-bit 32x16 layout): lane l holds column N=l%16, elements
  // are K {0..7,16..23} (lane<16) or {8..15,24..31} (lane>=16), same as A.
  // Held in VGPRs for the whole kernel.
  union BU { v16bf v; uint4 u[2]; };
  BU b0, b1;
  const uint4* wrow =
      reinterpret_cast<const uint4*>(wb + (uint32_t)(r * 32 + (lane & 15)) * 32u + (uint32_t)(lane >> 4) * 8u);
  b0.u[0] = wrow[0];  b0.u[1] = wrow[2];   // columns  0..15
  b1.u[0] = wrow[64]; b1.u[1] = wrow[66];  // columns 16..31

  // A operand addressing (ISA 16-bit 16x32 layout): lanes l and l+16 share
  // row M=l; lane<16 supplies K {0..7,16..23}, lane>=16 supplies {8..15,24..31}.
  const int mhalf = lane >> 4;             // 0 or 1
  const int mrow  = lane & 15;
  const uint32_t koff = (uint32_t)mhalf * 8u;

  for (int t = waveId; t < nTiles; t += wavesPerRel) {
    const int base = e0 + (t << 4);

    // Gather A tile: 16 edges x 32 K of bf16; u32 offsets keep saddr form.
    const uint32_t s = (uint32_t)src[base + mrow];
    const uint32_t xoff = s * 32u + koff;
    union AU { v16bf v; uint4 u[2]; } a;
    a.u[0] = *reinterpret_cast<const uint4*>(xb + xoff);        // K koff..+7
    a.u[1] = *reinterpret_cast<const uint4*>(xb + xoff + 16u);  // K koff+16..+23

    v8f c0 = {}; v8f c1 = {};
    c0 = __builtin_amdgcn_wmma_f32_16x16x32_bf16(false, a.v, false, b0.v,
                                                 (short)0, c0, false, false);
    c1 = __builtin_amdgcn_wmma_f32_16x16x32_bf16(false, a.v, false, b1.v,
                                                 (short)0, c1, false, false);

    // C layout: VGPR i, lane<16 -> row i, col lane; lane>=16 -> row i+8.
    const uint32_t nloc  = (uint32_t)mrow;
    const int      mbase = mhalf * 8;
#pragma unroll
    for (int i = 0; i < 8; ++i) {
      const uint32_t d = (uint32_t)dst[base + mbase + i];
      const uint32_t o = d * 32u + nloc;   // u32 element offset -> saddr atomics
      atomicAdd(acc + o,       c0[i]);
      atomicAdd(acc + o + 16u, c1[i]);
    }
  }
}

// ---- mean + ReLU, write output columns, re-emit bf16 for next layer --------
__global__ void finalize_kernel(const float* __restrict__ acc,
                                const float* __restrict__ deg,
                                float* __restrict__ out,
                                uint16_t* __restrict__ hb,
                                int colBase) {
  int i = blockIdx.x * blockDim.x + threadIdx.x;
  if (i < NNODES * DOUT) {
    int node = i >> 5, n = i & 31;
    float d = deg[node]; d = d > 1.0f ? d : 1.0f;
    float v = acc[i] / d;
    v = v > 0.0f ? v : 0.0f;
    out[node * OUTDIM + colBase + n] = v;
    hb[i] = f2bf(v);
  }
}

extern "C" void kernel_launch(void* const* d_in, const int* in_sizes, int n_in,
                              void* d_out, int out_size, void* d_ws, size_t ws_size,
                              hipStream_t stream) {
  (void)in_sizes; (void)n_in; (void)out_size; (void)ws_size;

  const float* x        = (const float*)d_in[0];
  const int*   edge_idx = (const int*)  d_in[1];
  // d_in[2] = edge_type (implicit in range_list; unused)
  const int*   rangel   = (const int*)  d_in[3];
  const float* basis0   = (const float*)d_in[4];
  const float* att0     = (const float*)d_in[5];
  const float* basis1   = (const float*)d_in[6];
  const float* att1     = (const float*)d_in[7];
  float* out = (float*)d_out;

  const int* src = edge_idx;
  const int* dst = edge_idx + NEDGES;

  // workspace carve-out (~20 MB)
  char* ws = (char*)d_ws;
  size_t off = 0;
  auto carve = [&](size_t bytes) {
    void* p = ws + off;
    off = (off + bytes + 255) & ~(size_t)255;
    return p;
  };
  float*    deg = (float*)   carve((size_t)NNODES * 4);
  float*    acc = (float*)   carve((size_t)NNODES * 32 * 4);
  uint16_t* xb  = (uint16_t*)carve((size_t)NNODES * 32 * 2);
  uint16_t* wb  = (uint16_t*)carve((size_t)NREL * 32 * 32 * 2);

  hipMemsetAsync(deg, 0, (size_t)NNODES * 4, stream);
  deg_kernel<<<(NEDGES + 255) / 256, 256, 0, stream>>>(dst, deg);
  copy_convert_kernel<<<(NNODES * 32 + 255) / 256, 256, 0, stream>>>(x, out, xb);

  dim3 egrid(64, NREL);

  // ---- layer 0 ----
  wmat_kernel<<<NREL, 1024, 0, stream>>>(att0, basis0, wb);
  hipMemsetAsync(acc, 0, (size_t)NNODES * 32 * 4, stream);
  edge_wmma_kernel<<<egrid, 256, 0, stream>>>(xb, wb, src, dst, rangel, acc);
  finalize_kernel<<<(NNODES * 32 + 255) / 256, 256, 0, stream>>>(acc, deg, out, xb, 32);

  // ---- layer 1 (reads bf16 h written by finalize) ----
  wmat_kernel<<<NREL, 1024, 0, stream>>>(att1, basis1, wb);
  hipMemsetAsync(acc, 0, (size_t)NNODES * 32 * 4, stream);
  edge_wmma_kernel<<<egrid, 256, 0, stream>>>(xb, wb, src, dst, rangel, acc);
  finalize_kernel<<<(NNODES * 32 + 255) / 256, 256, 0, stream>>>(acc, deg, out, xb, 64);
}
